// MultiHeadLatentAttention_50302656970912
// MI455X (gfx1250) — matrix operand order repaired and checked by reading, not compiled, not run on hardware
//
#include <hip/hip_runtime.h>
#include <hip/hip_bf16.h>

typedef __bf16 bf16;
typedef __attribute__((ext_vector_type(16))) __bf16 v16bf;
typedef __attribute__((ext_vector_type(8)))  __bf16 v8bf;
typedef __attribute__((ext_vector_type(8)))  float  v8f;
typedef __attribute__((ext_vector_type(4)))  unsigned int v4u;
typedef __attribute__((ext_vector_type(4)))  int v4i;
typedef __attribute__((ext_vector_type(8)))  int v8i;

constexpr int cB    = 2;
constexpr int cS    = 2048;
constexpr int cHID  = 2048;
constexpr int cNH   = 16;
constexpr int cNOPE = 128;
constexpr int cVD   = 128;
constexpr int cQD   = 192;   // NOPE + ROPE_D
constexpr int cLORA = 512;

// Build a 16-element bf16 fragment from two contiguous 8-element runs.
__device__ inline v16bf make_frag(const bf16* p0, const bf16* p1) {
  v8bf lo = *(const v8bf*)p0;
  v8bf hi = *(const v8bf*)p1;
  return __builtin_shufflevector(lo, hi, 0,1,2,3,4,5,6,7,8,9,10,11,12,13,14,15);
}

__device__ inline v8f wmma_bf16(v16bf a, v16bf b, v8f c) {
  return __builtin_amdgcn_wmma_f32_16x16x32_bf16(
      /*neg_a=*/false, a, /*neg_b=*/false, b,
      /*c_mod=*/(short)0, c, /*reuse_a=*/false, /*reuse_b=*/false);
}

// ---------------------------------------------------------------------------
// TDM: DMA a 64-row x 32-col bf16 tile (row stride = K elems) from global into
// LDS at lds_byte_off, padding each 16-DWORD row with 4 DWORDs so the LDS
// layout has a 40-element row stride (matches the fragment addressing below).
// D# layout per CDNA5 ISA.
// ---------------------------------------------------------------------------
__device__ inline void tdm_load_tile_bf16(const bf16* gsrc, unsigned lds_byte_off,
                                          int row_stride_elems) {
  unsigned long long ga = (unsigned long long)(size_t)gsrc;
  v4u g0;
  g0.x = 1u;                                               // count=1 (valid), user mode
  g0.y = lds_byte_off;                                     // lds_addr [63:32]
  g0.z = (unsigned)(ga & 0xffffffffu);                     // global_addr lo
  g0.w = (unsigned)((ga >> 32) & 0x01ffffffu) | (2u << 30);// global_addr hi | type=2
  v8i g1;
  g1[0] = (1 << 16)   // data_size = 2 bytes
        | (1 << 20)   // pad_enable
        | (3 << 22)   // pad_interval: 16 DWORDs (one 32-elem bf16 row)
        | (3 << 25);  // pad_amount: 4 DWORDs (8 bf16 elems) -> stride 40 elems
  g1[1] = (32 << 16);                 // tensor_dim0 = 32
  g1[2] = (64 << 16);                 // tensor_dim1 = 64
  g1[3] = (32 << 16);                 // tile_dim0 = 32
  g1[4] = 64;                         // tile_dim1 = 64, tile_dim2 = 0 (2D)
  g1[5] = row_stride_elems;           // tensor_dim0_stride lo32 (data_size units)
  g1[6] = 0;
  g1[7] = 0;
  v4i z4 = {0, 0, 0, 0};
#if defined(__clang_major__) && __clang_major__ >= 23
  v8i z8 = {0, 0, 0, 0, 0, 0, 0, 0};
  __builtin_amdgcn_tensor_load_to_lds(g0, g1, z4, z4, z8, 0);
#else
  __builtin_amdgcn_tensor_load_to_lds(g0, g1, z4, z4, 0);
#endif
}

// ---------------------------------------------------------------------------
// Tiled GEMM:  C[M,N] = A[M,K] (f32 or bf16, row-major) * B[K,N] (f32, row-major)
// MODE 0: store bf16 row-major          (TC = bf16)
// MODE 1: store bf16 as V^T [B,NH,VD,S] (TC = bf16; N interpreted as NH*VD)
// MODE 2: store f32 row-major           (TC = float)
// Block: 128 threads = 4 waves; block tile 64x64; K step 32.
// bf16 A-operand: tile staged by the Tensor Data Mover (async, TENSORcnt).
// ---------------------------------------------------------------------------
template<int MODE, typename TA, typename TC>
__global__ __launch_bounds__(128)
void gemm_wmma_kernel(const TA* __restrict__ A, const float* __restrict__ B,
                      TC* __restrict__ C, int M, int N, int K)
{
  constexpr int AST = 40;                 // padded LDS row stride (bf16 elems)
  constexpr int BST = 40;
  constexpr bool useTDM = (sizeof(TA) == 2);   // bf16 A: pure DMA, no convert
  alignas(16) __shared__ bf16 As[64 * AST];
  alignas(16) __shared__ bf16 Bs[64 * BST];

  const int tid  = threadIdx.x;
  const int n0   = blockIdx.x * 64;
  const int m0   = blockIdx.y * 64;
  const int wave = tid >> 5;
  const int lane = tid & 31;
  const int lm   = lane & 15;
  const int hi   = lane >> 4;
  const unsigned as_lds = (unsigned)(size_t)(void*)&As[0];  // LDS aperture: addr[31:0]

  v8f acc[4] = {};

  for (int k0 = 0; k0 < K; k0 += 32) {
    // Stage A tile 64x32
    if constexpr (useTDM) {
      if (wave == 0)
        tdm_load_tile_bf16((const bf16*)A + (size_t)m0 * K + k0, as_lds, K);
    } else {
      for (int i = tid; i < 64 * 32; i += 128) {
        int r = i >> 5, c = i & 31;
        As[r * AST + c] = (bf16)(float)A[(size_t)(m0 + r) * K + k0 + c];
      }
    }
    // Stage B tile 32x64, transposed -> Bs[n][k] so fragments read contiguous K
    for (int i = tid; i < 64 * 32; i += 128) {
      int kk = i >> 6, n = i & 63;
      Bs[n * BST + kk] = (bf16)B[(size_t)(k0 + kk) * N + n0 + n];
    }
    if (k0 + 32 < K) {                    // global_prefetch_b8 of next tiles
      __builtin_prefetch(&B[(size_t)(k0 + 32) * N + n0 + (tid & 63)], 0, 1);
      __builtin_prefetch(&A[(size_t)(m0 + (tid >> 1)) * K + k0 + 32], 0, 1);
    }
    if constexpr (useTDM)
      __builtin_amdgcn_s_wait_tensorcnt(0);
    __syncthreads();

    // A fragment (16x32): lane row = lm, 16 contiguous K at 16*hi
    const bf16* Ap = &As[(wave * 16 + lm) * AST + hi * 16];
    v16bf af = make_frag(Ap, Ap + 8);

    #pragma unroll
    for (int ns = 0; ns < 4; ++ns) {
      // B fragment (32x16): lane col = lm, 16 contiguous K at 16*hi
      const bf16* Bp = &Bs[(ns * 16 + lm) * BST + hi * 16];
      v16bf bfr = make_frag(Bp, Bp + 8);
      acc[ns] = wmma_bf16(af, bfr, acc[ns]);
    }
    __syncthreads();
  }

  // C fragment layout: vgpr i -> row i + 8*hi, col = lm (per 16-col subtile)
  #pragma unroll
  for (int ns = 0; ns < 4; ++ns) {
    #pragma unroll
    for (int i = 0; i < 8; ++i) {
      int gm = m0 + wave * 16 + i + 8 * hi;
      int gn = n0 + ns * 16 + lm;
      float v = acc[ns][i];
      if constexpr (MODE == 0) {
        C[(size_t)gm * N + gn] = (TC)(bf16)v;
      } else if constexpr (MODE == 1) {
        int b = gm / cS, s = gm % cS;
        int h = gn / cVD, f = gn % cVD;
        C[((size_t)(b * cNH + h) * cVD + f) * cS + s] = (TC)(bf16)v;
      } else {
        C[(size_t)gm * N + gn] = (TC)v;
      }
    }
  }
}

// ---------------------------------------------------------------------------
// RoPE in-place on bf16 [B,S,NH,QD]; rotates dims [NOPE, NOPE+64).
// One thread per rotation pair (j, j+32).
// ---------------------------------------------------------------------------
__global__ __launch_bounds__(256)
void rope_kernel(bf16* __restrict__ x)
{
  int t = blockIdx.x * 256 + threadIdx.x;
  int j = t & 31;
  int h = (t >> 5) & (cNH - 1);
  int s = (t >> 9) & (cS - 1);
  int b = t >> 20;
  bf16* p = x + ((size_t)((b * cS + s) * cNH + h)) * cQD + cNOPE;
  float x1 = (float)p[j];
  float x2 = (float)p[j + 32];
  float inv = __expf(-0.28782314f * (float)j);   // ln(10000)/32
  float fr  = (float)s * inv;
  float sn, c;
  __sincosf(fr, &sn, &c);
  p[j]      = (bf16)(x1 * c - x2 * sn);
  p[j + 32] = (bf16)(x2 * c + x1 * sn);
}

// ---------------------------------------------------------------------------
// Flash attention: 1 wave per (b, h, 16-query strip); key tile = 32.
// q,k: bf16 [B,S,NH,QD] (post-RoPE).  vt: bf16 [B,NH,VD,S].  o: bf16 [B,S,NH,VD].
// ---------------------------------------------------------------------------
__global__ __launch_bounds__(32)
void mla_attn_kernel(const bf16* __restrict__ q, const bf16* __restrict__ k,
                     const bf16* __restrict__ vt, bf16* __restrict__ o)
{
  alignas(16) __shared__ bf16 Pt[16 * 40];  // P tile staging (C-layout -> A-layout)
  const int lane = threadIdx.x & 31;
  const int lm = lane & 15, hi = lane >> 4;
  const int q0 = blockIdx.x * 16;
  const int h  = blockIdx.y;
  const int b  = blockIdx.z;
  const float scale = 0.07216878f;  // 1/sqrt(192)

  // Q strip: 6 A-fragments covering QD=192
  v16bf qf[6];
  {
    const bf16* qp = q + ((size_t)((b * cS + q0 + lm) * cNH + h)) * cQD + 16 * hi;
    #pragma unroll
    for (int kk = 0; kk < 6; ++kk)
      qf[kk] = make_frag(qp + 32 * kk, qp + 32 * kk + 8);
  }

  v8f oacc[8] = {};
  float mrow[8], lrow[8];
  #pragma unroll
  for (int i = 0; i < 8; ++i) { mrow[i] = -1e30f; lrow[i] = 0.f; }

  for (int kt = 0; kt < cS / 32; ++kt) {
    v8f s0 = {}, s1 = {};
    // K fragments: lane col = key index; 16 contiguous feature dims per lane
    const bf16* kp  = k + ((size_t)((b * cS + kt * 32 + lm) * cNH + h)) * cQD + 16 * hi;
    const bf16* kp2 = kp + (size_t)16 * cNH * cQD;   // keys +16
    #pragma unroll
    for (int kk = 0; kk < 6; ++kk) {
      v16bf b0 = make_frag(kp  + 32 * kk, kp  + 32 * kk + 8);
      v16bf b1 = make_frag(kp2 + 32 * kk, kp2 + 32 * kk + 8);
      s0 = wmma_bf16(qf[kk], b0, s0);
      s1 = wmma_bf16(qf[kk], b1, s1);
    }

    // Online softmax over the 32 keys of this tile
    float p0[8], p1[8];
    #pragma unroll
    for (int i = 0; i < 8; ++i) {
      float a = s0[i] * scale;
      float c = s1[i] * scale;
      float rm = fmaxf(a, c);
      #pragma unroll
      for (int m = 8; m >= 1; m >>= 1)
        rm = fmaxf(rm, __shfl_xor(rm, m, 32));
      float mn    = fmaxf(mrow[i], rm);
      float alpha = __expf(mrow[i] - mn);
      float e0 = __expf(a - mn);
      float e1 = __expf(c - mn);
      float rs = e0 + e1;
      #pragma unroll
      for (int m = 8; m >= 1; m >>= 1)
        rs += __shfl_xor(rs, m, 32);
      lrow[i] = lrow[i] * alpha + rs;
      mrow[i] = mn;
      p0[i] = e0; p1[i] = e1;
      #pragma unroll
      for (int j = 0; j < 8; ++j) oacc[j][i] *= alpha;
    }

    // P (16x32) C-layout -> LDS -> A-layout bf16 fragment
    __syncthreads();
    #pragma unroll
    for (int i = 0; i < 8; ++i) {
      int r = i + 8 * hi;
      Pt[r * 40 + lm]      = (bf16)p0[i];
      Pt[r * 40 + 16 + lm] = (bf16)p1[i];
    }
    __syncthreads();
    const bf16* Pp = &Pt[lm * 40 + 16 * hi];
    v16bf pf = make_frag(Pp, Pp + 8);

    // O += P @ V : V^T layout gives contiguous 16-key runs per lane
    const bf16* vp = vt + ((size_t)(b * cNH + h) * cVD) * cS + (size_t)(kt * 32 + 16 * hi);
    #pragma unroll
    for (int j = 0; j < 8; ++j) {
      const bf16* vpp = vp + (size_t)(j * 16 + lm) * cS;
      v16bf vf = make_frag(vpp, vpp + 8);
      oacc[j] = wmma_bf16(pf, vf, oacc[j]);
    }
  }

  // Normalize and store [B,S,NH,VD]
  #pragma unroll
  for (int j = 0; j < 8; ++j) {
    #pragma unroll
    for (int i = 0; i < 8; ++i) {
      int r = q0 + i + 8 * hi;
      int f = j * 16 + lm;
      o[((size_t)(b * cS + r) * cNH + h) * cVD + f] = (bf16)(oacc[j][i] / lrow[i]);
    }
  }
}

// ---------------------------------------------------------------------------
extern "C" void kernel_launch(void* const* d_in, const int* in_sizes, int n_in,
                              void* d_out, int out_size, void* d_ws, size_t ws_size,
                              hipStream_t stream)
{
  (void)in_sizes; (void)n_in; (void)out_size; (void)ws_size;
  const float* hs  = (const float*)d_in[0];
  const float* Wq  = (const float*)d_in[1];
  const float* Wkv = (const float*)d_in[2];
  const float* Wku = (const float*)d_in[3];
  const float* Wvu = (const float*)d_in[4];
  const float* Wo  = (const float*)d_in[5];
  float* out = (float*)d_out;

  const int M = cB * cS;                         // 4096 tokens

  bf16* q_bf   = (bf16*)d_ws;                    // [M, NH*QD]
  bf16* k_bf   = q_bf   + (size_t)M * cNH * cQD; // [M, NH*QD]
  bf16* ckv_bf = k_bf   + (size_t)M * cNH * cQD; // [M, LORA]
  bf16* v_t    = ckv_bf + (size_t)M * cLORA;     // [B,NH,VD,S]
  bf16* at_bf  = v_t    + (size_t)M * cNH * cVD; // [M, NH*VD]

  dim3 blk(128);
  gemm_wmma_kernel<0, float, bf16><<<dim3((cNH * cQD) / 64, M / 64), blk, 0, stream>>>(
      hs, Wq, q_bf, M, cNH * cQD, cHID);
  gemm_wmma_kernel<0, float, bf16><<<dim3(cLORA / 64, M / 64), blk, 0, stream>>>(
      hs, Wkv, ckv_bf, M, cLORA, cHID);
  gemm_wmma_kernel<0, bf16, bf16><<<dim3((cNH * cQD) / 64, M / 64), blk, 0, stream>>>(
      ckv_bf, Wku, k_bf, M, cNH * cQD, cLORA);
  gemm_wmma_kernel<1, bf16, bf16><<<dim3((cNH * cVD) / 64, M / 64), blk, 0, stream>>>(
      ckv_bf, Wvu, v_t, M, cNH * cVD, cLORA);

  int ropeN = cB * cS * cNH * 32;
  rope_kernel<<<ropeN / 256, 256, 0, stream>>>(q_bf);
  rope_kernel<<<ropeN / 256, 256, 0, stream>>>(k_bf);

  mla_attn_kernel<<<dim3(cS / 16, cNH, cB), 32, 0, stream>>>(q_bf, k_bf, v_t, at_bf);

  gemm_wmma_kernel<2, bf16, float><<<dim3(cHID / 64, M / 64), blk, 0, stream>>>(
      at_bf, Wo, out, M, cHID, cNH * cVD);
}
